// SelfAttention_45853070852655
// MI455X (gfx1250) — matrix operand order repaired and checked by reading, not compiled, not run on hardware
//
#include <hip/hip_runtime.h>
#include <hip/hip_bf16.h>

#define BB   2
#define SS   2048
#define HSZ  2048
#define NH   16
#define HD   128
#define BSR  (BB * SS)          // 4096 rows

typedef __attribute__((ext_vector_type(16))) __bf16 v16bf;
typedef __attribute__((ext_vector_type(8)))  __bf16 v8bf;
typedef __attribute__((ext_vector_type(8)))  float  v8f;

static __device__ __forceinline__ v8f wmma_bf16(v16bf a, v16bf b, v8f c) {
    return __builtin_amdgcn_wmma_f32_16x16x32_bf16(
        /*neg_a=*/false, a, /*neg_b=*/false, b,
        /*c_mod=*/(short)0, c, /*reuse_a=*/false, /*reuse_b=*/false);
}

// ---------------------------------------------------------------- converts
__global__ __launch_bounds__(256, 1)
void f32_to_bf16_kernel(const float* __restrict__ x,
                        __bf16* __restrict__ y, size_t n) {
    size_t i = (size_t)blockIdx.x * blockDim.x + threadIdx.x;
    size_t stride = (size_t)gridDim.x * blockDim.x;
    for (; i < n; i += stride) y[i] = (__bf16)x[i];
}

// ---------------------------------------------------------------- RoPE
// X: (BSR, HSZ) f32 laid out as [b*S+s][h*HD+d]; writes bf16.
__global__ __launch_bounds__(256, 1)
void rope_kernel(const float* __restrict__ X,
                 const float* __restrict__ cosT,
                 const float* __restrict__ sinT,
                 __bf16* __restrict__ Y) {
    const size_t n = (size_t)BSR * HSZ;
    size_t i = (size_t)blockIdx.x * blockDim.x + threadIdx.x;
    size_t stride = (size_t)gridDim.x * blockDim.x;
    for (; i < n; i += stride) {
        int col = (int)(i % HSZ);
        size_t row = i / HSZ;
        int s = (int)(row % SS);
        int d = col % HD;
        float c  = cosT[(size_t)s * HD + d];
        float sn = sinT[(size_t)s * HD + d];
        float x  = X[i];
        float other = (d < HD / 2) ? -X[i + HD / 2] : X[i - HD / 2];
        Y[i] = (__bf16)(x * c + other * sn);
    }
}

// ---------------------------------------------------------------- NT GEMM
// C(MxN) = A(MxK) * Bw(NxK)^T ; one wave computes a 32x64 tile
// (two 16-row sub-tiles sharing the four B fragments).
// Grid must supply exactly (M/32)*(N/64) waves.
// MODE 0: f32 row-major   MODE 1: bf16 row-major
// MODE 2: bf16, written transposed per head: Vt[(b*NH+h)*HD + d][s]
template <int MODE>
__global__ __launch_bounds__(256, 1)
void gemm_nt_kernel(const __bf16* __restrict__ A,
                    const __bf16* __restrict__ Bw,
                    float* __restrict__ Cf,
                    __bf16* __restrict__ Cb,
                    int M, int N, int K) {
    const int lane = threadIdx.x & 31;
    const int wave = blockIdx.x * (blockDim.x >> 5) + (threadIdx.x >> 5);
    const int nTiles = N / 64;
    const int m0 = (wave / nTiles) * 32;
    const int n0 = (wave % nTiles) * 64;
    const int half = lane >> 4;
    const int l16  = lane & 15;

    v8f acc[2][4] = {};
    const size_t arow0 = (size_t)(m0 + l16) * K;
    const size_t arow1 = (size_t)(m0 + 16 + l16) * K;
    const __bf16* bp0 = Bw + (size_t)(n0 +  0 + l16) * K + half * 16;
    const __bf16* bp1 = Bw + (size_t)(n0 + 16 + l16) * K + half * 16;
    const __bf16* bp2 = Bw + (size_t)(n0 + 32 + l16) * K + half * 16;
    const __bf16* bp3 = Bw + (size_t)(n0 + 48 + l16) * K + half * 16;

    for (int k = 0; k < K; k += 32) {
        union { v16bf v; v8bf h[2]; } au0, au1;
        au0.h[0] = *(const v8bf*)(A + arow0 + k + half * 16);
        au0.h[1] = *(const v8bf*)(A + arow0 + k + half * 16 + 8);
        au1.h[0] = *(const v8bf*)(A + arow1 + k + half * 16);
        au1.h[1] = *(const v8bf*)(A + arow1 + k + half * 16 + 8);

        v16bf b0 = *(const v16bf*)(bp0 + k);
        v16bf b1 = *(const v16bf*)(bp1 + k);
        v16bf b2 = *(const v16bf*)(bp2 + k);
        v16bf b3 = *(const v16bf*)(bp3 + k);

        acc[0][0] = wmma_bf16(au0.v, b0, acc[0][0]);
        acc[0][1] = wmma_bf16(au0.v, b1, acc[0][1]);
        acc[0][2] = wmma_bf16(au0.v, b2, acc[0][2]);
        acc[0][3] = wmma_bf16(au0.v, b3, acc[0][3]);
        acc[1][0] = wmma_bf16(au1.v, b0, acc[1][0]);
        acc[1][1] = wmma_bf16(au1.v, b1, acc[1][1]);
        acc[1][2] = wmma_bf16(au1.v, b2, acc[1][2]);
        acc[1][3] = wmma_bf16(au1.v, b3, acc[1][3]);
    }

    // C layout: lane holds (M = v + 8*half, N = l16) in VGPR v.
#pragma unroll
    for (int r = 0; r < 2; ++r) {
#pragma unroll
        for (int t = 0; t < 4; ++t) {
#pragma unroll
            for (int v = 0; v < 8; ++v) {
                const int row = m0 + r * 16 + v + 8 * half;
                const int col = n0 + t * 16 + l16;
                float val = acc[r][t][v];
                if (MODE == 0) {
                    Cf[(size_t)row * N + col] = val;
                } else if (MODE == 1) {
                    Cb[(size_t)row * N + col] = (__bf16)val;
                } else {
                    // row = b*SS + s ; col = h*HD + d -> Vt[((b*NH+h)*HD+d)*SS + s]
                    const int b = row / SS, s = row % SS;
                    const int h = col / HD, d = col % HD;
                    Cb[((size_t)(b * NH + h) * HD + d) * SS + s] = (__bf16)val;
                }
            }
        }
    }
}

// ---------------------------------------------------------------- flash attention
// One wave per (b, h, 16-query tile). Grid supplies exactly B*NH*(S/16) waves.
// Vt is the per-head transposed V: Vt[(b*NH+h)*HD + d][s].
__global__ __launch_bounds__(256, 1)
void flash_attn_kernel(const __bf16* __restrict__ Qb,
                       const __bf16* __restrict__ Kb,
                       const __bf16* __restrict__ Vt,
                       __bf16* __restrict__ Ob) {
    __shared__ float plds[8][16][32];   // per-wave P-tile transpose buffer

    const int lane  = threadIdx.x & 31;
    const int wslot = threadIdx.x >> 5;
    const int wave  = blockIdx.x * (blockDim.x >> 5) + wslot;
    const int QT = SS / 16;
    const int qt = wave % QT;
    const int bh = wave / QT;          // b*NH + h
    const int h  = bh % NH;
    const int b  = bh / NH;
    const int q0 = qt * 16;
    const int half = lane >> 4;
    const int l16  = lane & 15;
    const float scale = 0.08838834764831845f;   // 1/sqrt(128)

    // Q A-fragments over HD=128 (4 chunks of 32)
    v16bf qfrag[4];
    const size_t qrow = ((size_t)(b * SS + q0 + l16)) * HSZ + (size_t)h * HD;
#pragma unroll
    for (int c = 0; c < 4; ++c) {
        union { v16bf v; v8bf hh[2]; } u;
        u.hh[0] = *(const v8bf*)(Qb + qrow + c * 32 + half * 16);
        u.hh[1] = *(const v8bf*)(Qb + qrow + c * 32 + half * 16 + 8);
        qfrag[c] = u.v;
    }

    float rowm[8], rowl[8];
#pragma unroll
    for (int v = 0; v < 8; ++v) { rowm[v] = -1e30f; rowl[v] = 0.0f; }
    v8f oacc[8] = {};   // 16x128 f32 output accumulator (8 N-tiles)

    const __bf16* kbase  = Kb + (size_t)(b * SS) * HSZ + (size_t)h * HD + half * 16;
    const __bf16* vtbase = Vt + (size_t)bh * HD * SS;

    for (int j = 0; j < q0 + 16; j += 32) {
        // scores: two 16x16 tiles (keys j..j+15 and j+16..j+31)
        v8f sc0 = {}, sc1 = {};
        const __bf16* kp0 = kbase + (size_t)(j + l16) * HSZ;
        const __bf16* kp1 = kbase + (size_t)(j + 16 + l16) * HSZ;
#pragma unroll
        for (int c = 0; c < 4; ++c) {
            sc0 = wmma_bf16(qfrag[c], *(const v16bf*)(kp0 + c * 32), sc0);
            sc1 = wmma_bf16(qfrag[c], *(const v16bf*)(kp1 + c * 32), sc1);
        }

        // causal mask + online softmax, per local row (M = v + 8*half)
#pragma unroll
        for (int v = 0; v < 8; ++v) {
            const int qi   = q0 + v + 8 * half;
            const int key0 = j + l16;
            const int key1 = j + 16 + l16;
            float f0 = (key0 <= qi) ? sc0[v] * scale : -1e30f;
            float f1 = (key1 <= qi) ? sc1[v] * scale : -1e30f;
            float x = fmaxf(f0, f1);
#pragma unroll
            for (int m = 1; m < 16; m <<= 1) x = fmaxf(x, __shfl_xor(x, m, 16));
            const float mnew  = fmaxf(rowm[v], x);
            const float alpha = __expf(rowm[v] - mnew);
            rowm[v] = mnew;
            float p0 = (key0 <= qi) ? __expf(f0 - mnew) : 0.0f;
            float p1 = (key1 <= qi) ? __expf(f1 - mnew) : 0.0f;
            float rs = p0 + p1;
#pragma unroll
            for (int m = 1; m < 16; m <<= 1) rs += __shfl_xor(rs, m, 16);
            rowl[v] = rowl[v] * alpha + rs;
#pragma unroll
            for (int nt = 0; nt < 8; ++nt) oacc[nt][v] *= alpha;
            // stage P for C-layout -> A-layout transpose
            plds[wslot][v + 8 * half][l16]      = p0;
            plds[wslot][v + 8 * half][16 + l16] = p1;
        }

        asm volatile("s_wait_dscnt 0" ::: "memory");

        // P as bf16 A-fragment (16 rows x 32 keys)
        union { v16bf v; __bf16 e[16]; } pu;
#pragma unroll
        for (int e = 0; e < 16; ++e)
            pu.e[e] = (__bf16)plds[wslot][l16][half * 16 + e];

        // PV: B fragment from head-transposed V -> one contiguous 32B load / tile
#pragma unroll
        for (int nt = 0; nt < 8; ++nt) {
            const __bf16* vp = vtbase + (size_t)(nt * 16 + l16) * SS + j + half * 16;
            oacc[nt] = wmma_bf16(pu.v, *(const v16bf*)vp, oacc[nt]);
        }
    }

    // epilogue: divide by row sums, write bf16
#pragma unroll
    for (int nt = 0; nt < 8; ++nt) {
#pragma unroll
        for (int v = 0; v < 8; ++v) {
            size_t row = (size_t)(b * SS + q0 + v + 8 * half);
            size_t col = (size_t)h * HD + nt * 16 + l16;
            Ob[row * HSZ + col] = (__bf16)(oacc[nt][v] / rowl[v]);
        }
    }
}

// ---------------------------------------------------------------- launcher
extern "C" void kernel_launch(void* const* d_in, const int* in_sizes, int n_in,
                              void* d_out, int out_size, void* d_ws, size_t ws_size,
                              hipStream_t stream) {
    const float* hs   = (const float*)d_in[0];
    const float* wq   = (const float*)d_in[1];
    const float* wk   = (const float*)d_in[2];
    const float* wv   = (const float*)d_in[3];
    const float* wo   = (const float*)d_in[4];
    const float* cosT = (const float*)d_in[5];
    const float* sinT = (const float*)d_in[6];
    float* out = (float*)d_out;

    char* p = (char*)d_ws;
    __bf16* Hb  = (__bf16*)p; p += (size_t)BSR * HSZ * 2;
    __bf16* Wqb = (__bf16*)p; p += (size_t)HSZ * HSZ * 2;
    __bf16* Wkb = (__bf16*)p; p += (size_t)HSZ * HSZ * 2;
    __bf16* Wvb = (__bf16*)p; p += (size_t)HSZ * HSZ * 2;
    __bf16* Wob = (__bf16*)p; p += (size_t)HSZ * HSZ * 2;
    float*  Qf  = (float*)p;  p += (size_t)BSR * HSZ * 4;
    float*  Kf  = (float*)p;  p += (size_t)BSR * HSZ * 4;
    __bf16* Qr  = (__bf16*)p; p += (size_t)BSR * HSZ * 2;
    __bf16* Kr  = (__bf16*)p; p += (size_t)BSR * HSZ * 2;
    __bf16* Vt  = (__bf16*)p; p += (size_t)BSR * HSZ * 2;
    __bf16* Ab  = (__bf16*)p; p += (size_t)BSR * HSZ * 2;

    const size_t nHS = (size_t)BSR * HSZ;
    const size_t nW  = (size_t)HSZ * HSZ;

    f32_to_bf16_kernel<<<4096, 256, 0, stream>>>(hs, Hb, nHS);
    f32_to_bf16_kernel<<<2048, 256, 0, stream>>>(wq, Wqb, nW);
    f32_to_bf16_kernel<<<2048, 256, 0, stream>>>(wk, Wkb, nW);
    f32_to_bf16_kernel<<<2048, 256, 0, stream>>>(wv, Wvb, nW);
    f32_to_bf16_kernel<<<2048, 256, 0, stream>>>(wo, Wob, nW);

    // (M/32)*(N/64) = 128*32 = 4096 waves; 8 waves/block -> 512 blocks
    gemm_nt_kernel<0><<<512, 256, 0, stream>>>(Hb, Wqb, Qf, nullptr, BSR, HSZ, HSZ);
    gemm_nt_kernel<0><<<512, 256, 0, stream>>>(Hb, Wkb, Kf, nullptr, BSR, HSZ, HSZ);
    gemm_nt_kernel<2><<<512, 256, 0, stream>>>(Hb, Wvb, nullptr, Vt, BSR, HSZ, HSZ);

    rope_kernel<<<4096, 256, 0, stream>>>(Qf, cosT, sinT, Qr);
    rope_kernel<<<4096, 256, 0, stream>>>(Kf, cosT, sinT, Kr);

    // B*NH*(S/16) = 4096 waves; 8 waves/block -> 512 blocks
    flash_attn_kernel<<<512, 256, 0, stream>>>(Qr, Kr, Vt, Ab);

    gemm_nt_kernel<0><<<512, 256, 0, stream>>>(Ab, Wob, out, nullptr, BSR, HSZ, HSZ);
}
